// FCE_f_20538533609520
// MI455X (gfx1250) — matrix operand order repaired and checked by reading, not compiled, not run on hardware
//
#include <hip/hip_runtime.h>
#include <math.h>

#define B_SZ 1024
#define D_SZ 1024
#define NSTEP 16
#define KTILES (D_SZ / 32)

#define BM 64      // batch rows per workgroup
#define BN 32      // cols per gate per workgroup
#define BK 32      // K-slab per iteration (== wmma f16 K)
#define LDP (BK + 8)  // padded LDS row stride (halves): 80B, 16B-aligned

#define GATE_STRIDE ((size_t)D_SZ * D_SZ)   // halves between gate weight blocks

typedef __attribute__((ext_vector_type(16))) _Float16 v16h;
typedef __attribute__((ext_vector_type(8)))  _Float16 v8h;
typedef __attribute__((ext_vector_type(8)))  float    v8f;

__device__ __forceinline__ float sigmoidf_(float x) {
    return 1.0f / (1.0f + __expf(-x));
}

__device__ __forceinline__ v16h make_frag(const _Float16* p0, const _Float16* p1) {
    v8h lo = *(const v8h*)p0;
    v8h hi = *(const v8h*)p1;
    return __builtin_shufflevector(lo, hi, 0,1,2,3,4,5,6,7,8,9,10,11,12,13,14,15);
}

__device__ __forceinline__ v8h pack8(float4 a, float4 b) {
    v8h p = { (_Float16)a.x, (_Float16)a.y, (_Float16)a.z, (_Float16)a.w,
              (_Float16)b.x, (_Float16)b.y, (_Float16)b.z, (_Float16)b.w };
    return p;
}

// ---------------------------------------------------------------------------
// One-time fp32 -> f16 weight pack (weights stay in this layout, L2-resident,
// for all 16 steps; B fragments load straight from global as b128 pairs).
// ---------------------------------------------------------------------------
__global__ __launch_bounds__(256) void pack_f16(const float* __restrict__ s,
                                                _Float16* __restrict__ d) {
    const size_t i = ((size_t)blockIdx.x * 256 + threadIdx.x) * 8;
    float4 a = *(const float4*)(s + i);
    float4 b = *(const float4*)(s + i + 4);
    *(v8h*)(d + i) = pack8(a, b);
}

// ---------------------------------------------------------------------------
// Recurrent step: gates = xproj + (h_prev + r) @ W_hh^T ; fused LSTM cell.
// 8 waves (4 M x 2 N); each wave holds one 16x16 f32 accumulator per gate.
// BOTH operands are software-pipelined one K-slab ahead: A through
// double-buffered LDS, B through a double set of register fragments, so the
// 4 WMMAs per slab issue back-to-back with no load waits in between.
// ---------------------------------------------------------------------------
__global__ __launch_bounds__(256) void lstm_step(
    const float* __restrict__ hprev, const float* __restrict__ rprev,
    const _Float16* __restrict__ W16, const float* __restrict__ xproj,
    float* __restrict__ cst,          const float* __restrict__ inp,
    float* __restrict__ hout)
{
    __shared__ __align__(16) _Float16 Ah[2][BM][LDP];

    const int tid  = threadIdx.x;
    const int lane = tid & 31;
    const int wave = tid >> 5;
    const int wm   = wave & 3;        // wave row block (16 rows)
    const int wn   = wave >> 2;       // wave col block (16 cols)
    const int lm   = lane & 15;
    const int hi   = lane >> 4;

    const int n0  = blockIdx.x * BN;
    const int bm0 = blockIdx.y * BM;

    // A staging: 256 threads cover 64 rows x 32 K (8 halves each)
    const int arow = tid >> 2;
    const int aseg = (tid & 3) * 8;

    const float* hp = hprev + (size_t)(bm0 + arow) * D_SZ + aseg;
    const float* rp = rprev + aseg;

    // per-lane B fragment base: lane = column N, K[8h..8h+7],K[16+8h..16+8h+7] per lane;
    // gate blocks become 24-bit immediate offsets (g * 2MB)
    const _Float16* wp = W16 + (size_t)(n0 + wn * 16 + lm) * D_SZ + hi * 8;

    // ---- prologue: stage K-slab 0 (A -> LDS buf0, B -> bf[0]) ----
    {
        float4 x0 = *(const float4*)(hp);
        float4 x1 = *(const float4*)(hp + 4);
        float4 r0 = *(const float4*)(rp);
        float4 r1 = *(const float4*)(rp + 4);
        float4 s0 = make_float4(x0.x + r0.x, x0.y + r0.y, x0.z + r0.z, x0.w + r0.w);
        float4 s1 = make_float4(x1.x + r1.x, x1.y + r1.y, x1.z + r1.z, x1.w + r1.w);
        *(v8h*)&Ah[0][arow][aseg] = pack8(s0, s1);
    }

    v16h bf[2][4];
    #pragma unroll
    for (int g = 0; g < 4; ++g)
        bf[0][g] = make_frag(wp + g * GATE_STRIDE, wp + g * GATE_STRIDE + 16);

    v8f acc[4] = {};

    #pragma unroll 2
    for (int kt = 0; kt < KTILES; ++kt) {
        __syncthreads();
        const int cur = kt & 1, nxt = cur ^ 1;
        const bool more = (kt + 1) < KTILES;

        // issue next slab's loads early (A floats + all 8 B b128s) —
        // consumed only at the NEXT iteration, hidden under this slab's WMMAs
        float4 x0, x1, r0, r1;
        if (more) {
            const float* h2 = hp + (kt + 1) * BK;
            const float* r2 = rp + (kt + 1) * BK;
            x0 = *(const float4*)(h2);
            x1 = *(const float4*)(h2 + 4);
            r0 = *(const float4*)(r2);
            r1 = *(const float4*)(r2 + 4);
            const _Float16* wk = wp + (kt + 1) * BK;
            #pragma unroll
            for (int g = 0; g < 4; ++g)
                bf[nxt][g] = make_frag(wk + g * GATE_STRIDE, wk + g * GATE_STRIDE + 16);
            if (kt + 2 < KTILES) __builtin_prefetch(hp + (kt + 2) * BK, 0, 3);
        }

        // A frag: lanes 0-15 -> K[0..7],K[16..23]; lanes 16-31 -> K[8..15],K[24..31]
        v16h af = make_frag(&Ah[cur][wm * 16 + lm][hi * 8],
                            &Ah[cur][wm * 16 + lm][16 + hi * 8]);

        #pragma unroll
        for (int g = 0; g < 4; ++g)
            acc[g] = __builtin_amdgcn_wmma_f32_16x16x32_f16(
                false, af, false, bf[cur][g], (short)0, acc[g], false, false);

        if (more) {
            float4 s0 = make_float4(x0.x + r0.x, x0.y + r0.y, x0.z + r0.z, x0.w + r0.w);
            float4 s1 = make_float4(x1.x + r1.x, x1.y + r1.y, x1.z + r1.z, x1.w + r1.w);
            *(v8h*)&Ah[nxt][arow][aseg] = pack8(s0, s1);
        }
    }

    // Fused LSTM cell epilogue. C/D layout: VGPR j -> M = j + 8*hi, N = lane%16.
    const int gn = n0 + wn * 16 + lm;
    #pragma unroll
    for (int j = 0; j < 8; ++j) {
        const int gb = bm0 + wm * 16 + hi * 8 + j;
        const size_t row4 = (size_t)gb * (4 * D_SZ);
        float xi = xproj[row4 + 0 * D_SZ + gn] + acc[0][j];
        float xf = xproj[row4 + 1 * D_SZ + gn] + acc[1][j];
        float xg = xproj[row4 + 2 * D_SZ + gn] + acc[2][j];
        float xo = xproj[row4 + 3 * D_SZ + gn] + acc[3][j];
        float iv = sigmoidf_(xi);
        float fv = sigmoidf_(xf);
        float gv = tanhf(xg);
        float ov = sigmoidf_(xo);
        const size_t idx = (size_t)gb * D_SZ + gn;
        float cn = fv * cst[idx] + iv * gv;
        cst[idx]  = cn;
        hout[idx] = ov * tanhf(cn) + inp[idx];   // h_k = h_cap + inputs
    }
}

// ---------------------------------------------------------------------------
// One-time projection: xproj = inputs @ W_ih^T + b_ih + b_hh (b_hh folded in)
// ---------------------------------------------------------------------------
__global__ __launch_bounds__(256) void xproj_gemm(
    const float* __restrict__ inp, const _Float16* __restrict__ W16,
    const float* __restrict__ bih, const float* __restrict__ bhh,
    float* __restrict__ xproj)
{
    __shared__ __align__(16) _Float16 Ah[2][BM][LDP];

    const int tid  = threadIdx.x;
    const int lane = tid & 31;
    const int wave = tid >> 5;
    const int wm   = wave & 3;
    const int wn   = wave >> 2;
    const int lm   = lane & 15;
    const int hi   = lane >> 4;

    const int n0  = blockIdx.x * BN;
    const int bm0 = blockIdx.y * BM;

    const int arow = tid >> 2;
    const int aseg = (tid & 3) * 8;

    const float* ap = inp + (size_t)(bm0 + arow) * D_SZ + aseg;
    const _Float16* wp = W16 + (size_t)(n0 + wn * 16 + lm) * D_SZ + hi * 8;

    {
        float4 x0 = *(const float4*)(ap);
        float4 x1 = *(const float4*)(ap + 4);
        *(v8h*)&Ah[0][arow][aseg] = pack8(x0, x1);
    }

    v16h bf[2][4];
    #pragma unroll
    for (int g = 0; g < 4; ++g)
        bf[0][g] = make_frag(wp + g * GATE_STRIDE, wp + g * GATE_STRIDE + 16);

    v8f acc[4] = {};

    #pragma unroll 2
    for (int kt = 0; kt < KTILES; ++kt) {
        __syncthreads();
        const int cur = kt & 1, nxt = cur ^ 1;
        const bool more = (kt + 1) < KTILES;

        float4 x0, x1;
        if (more) {
            const float* a2 = ap + (kt + 1) * BK;
            x0 = *(const float4*)(a2);
            x1 = *(const float4*)(a2 + 4);
            const _Float16* wk = wp + (kt + 1) * BK;
            #pragma unroll
            for (int g = 0; g < 4; ++g)
                bf[nxt][g] = make_frag(wk + g * GATE_STRIDE, wk + g * GATE_STRIDE + 16);
        }

        v16h af = make_frag(&Ah[cur][wm * 16 + lm][hi * 8],
                            &Ah[cur][wm * 16 + lm][16 + hi * 8]);

        #pragma unroll
        for (int g = 0; g < 4; ++g)
            acc[g] = __builtin_amdgcn_wmma_f32_16x16x32_f16(
                false, af, false, bf[cur][g], (short)0, acc[g], false, false);

        if (more)
            *(v8h*)&Ah[nxt][arow][aseg] = pack8(x0, x1);
    }

    const int gn = n0 + wn * 16 + lm;
    #pragma unroll
    for (int j = 0; j < 8; ++j) {
        const int gb = bm0 + wm * 16 + hi * 8 + j;
        const size_t row4 = (size_t)gb * (4 * D_SZ);
        #pragma unroll
        for (int g = 0; g < 4; ++g)
            xproj[row4 + g * D_SZ + gn] = acc[g][j] + bih[g * D_SZ + gn] + bhh[g * D_SZ + gn];
    }
}

// ---------------------------------------------------------------------------
// Content attention: per column d, online softmax over the batch axis of
// (h_prev + r) * memory, then r_out[d] = sum_b attn * memory. Column-parallel
// threads -> every row access is fully coalesced.
// ---------------------------------------------------------------------------
__global__ __launch_bounds__(256) void attention_step(
    const float* __restrict__ hprev, const float* __restrict__ rprev,
    const float* __restrict__ mem,   float* __restrict__ rout)
{
    const int d = blockIdx.x * blockDim.x + threadIdx.x;
    const float rv = rprev[d];
    float mx = -INFINITY, s = 0.0f, w = 0.0f;
    for (int b = 0; b < B_SZ; ++b) {
        const float mb = mem[(size_t)b * D_SZ + d];
        const float v  = (hprev[(size_t)b * D_SZ + d] + rv) * mb;
        const float mn = fmaxf(mx, v);
        const float sc = __expf(mx - mn);   // first iter: exp(-inf)=0
        const float e  = __expf(v - mn);
        s = s * sc + e;
        w = w * sc + e * mb;
        mx = mn;
    }
    rout[d] = w / s;
}

extern "C" void kernel_launch(void* const* d_in, const int* in_sizes, int n_in,
                              void* d_out, int out_size, void* d_ws, size_t ws_size,
                              hipStream_t stream) {
    const float* inputs = (const float*)d_in[0];
    const float* memory = (const float*)d_in[1];
    const float* W_ih   = (const float*)d_in[2];
    const float* W_hh   = (const float*)d_in[3];
    const float* b_ih   = (const float*)d_in[4];
    const float* b_hh   = (const float*)d_in[5];
    float* out = (float*)d_out;

    const size_t M1 = (size_t)1024 * 1024;
    float* ws    = (float*)d_ws;
    float* xproj = ws;                 // 4M floats
    float* h0    = xproj + 4 * M1;     // 1M
    float* h1    = h0 + M1;            // 1M
    float* cbuf  = h1 + M1;            // 1M
    float* r0    = cbuf + M1;          // 1K
    float* r1    = r0 + 1024;          // 1K
    _Float16* Whh16 = (_Float16*)(r1 + 1024);   // 4M halves
    _Float16* Wih16 = Whh16 + 4 * M1;           // 4M halves

    // zero initial carry (h0, h1, c, r0, r1) — ws is poisoned by the harness
    hipMemsetAsync(h0, 0, (3 * M1 + 2048) * sizeof(float), stream);

    // one-time f16 weight packs (stay L2-resident for the whole recurrence)
    pack_f16<<<(4 * M1) / (256 * 8), 256, 0, stream>>>(W_hh, Whh16);
    pack_f16<<<(4 * M1) / (256 * 8), 256, 0, stream>>>(W_ih, Wih16);

    dim3 grid(D_SZ / BN, B_SZ / BM);   // 32 x 16
    xproj_gemm<<<grid, 256, 0, stream>>>(inputs, Wih16, b_ih, b_hh, xproj);

    float* hb[2] = {h0, h1};
    float* rb[2] = {r0, r1};
    for (int s = 0; s < NSTEP; ++s) {
        const int cur = s & 1, nxt = cur ^ 1;
        if (s < NSTEP - 1)   // r from the last step is never consumed
            attention_step<<<D_SZ / 256, 256, 0, stream>>>(hb[cur], rb[cur], memory, rb[nxt]);
        float* ho = (s == NSTEP - 1) ? out : hb[nxt];    // scan output is h_k (pre-r)
        lstm_step<<<grid, 256, 0, stream>>>(hb[cur], rb[cur], Whh16, xproj, cbuf, inputs, ho);
    }
}
